// Encoder1DBlock_57269093925017
// MI455X (gfx1250) — matrix operand order repaired and checked by reading, not compiled, not run on hardware
//
#include <hip/hip_runtime.h>
#include <hip/hip_bf16.h>

typedef __bf16 bf16;
typedef __attribute__((ext_vector_type(16))) __bf16 v16bf;
typedef __attribute__((ext_vector_type(8)))  float  v8f;

// ---------------------------------------------------------------------------
// CDNA5 async global->LDS copy (ASYNCcnt-tracked, bypasses VGPRs)
// ---------------------------------------------------------------------------
__device__ __forceinline__ void async_ld_b128(unsigned lds_addr, const void* gaddr) {
    asm volatile("global_load_async_to_lds_b128 %0, %1, off"
                 :: "v"(lds_addr), "v"(gaddr) : "memory");
}
__device__ __forceinline__ void wait_async0() {
    asm volatile("s_wait_asynccnt 0" ::: "memory");
}

// ---------------------------------------------------------------------------
// fp32 -> bf16 convert (weights; re-done every call for determinism)
// ---------------------------------------------------------------------------
__global__ __launch_bounds__(256)
void cvt_f32_bf16(const float* __restrict__ in, bf16* __restrict__ out, int n) {
    int i = blockIdx.x * 256 + threadIdx.x;
    if (i < n) out[i] = (bf16)in[i];
}

// ---------------------------------------------------------------------------
// block reduction helper (256 threads = 8 waves of 32)
// ---------------------------------------------------------------------------
__device__ __forceinline__ float blk_sum(float v, float* red) {
#pragma unroll
    for (int o = 16; o > 0; o >>= 1) v += __shfl_xor(v, o, 32);
    int w = threadIdx.x >> 5;
    if ((threadIdx.x & 31) == 0) red[w] = v;
    __syncthreads();
    if (threadIdx.x == 0) {
        float s = 0.f;
#pragma unroll
        for (int i = 0; i < 8; i++) s += red[i];
        red[8] = s;
    }
    __syncthreads();
    float r = red[8];
    __syncthreads();
    return r;
}

// ---------------------------------------------------------------------------
// LayerNorm kernels, D = 1024, one row per 256-thread block, 4 elems/thread
// ---------------------------------------------------------------------------
__global__ __launch_bounds__(256)
void ln_cast_kernel(const float* __restrict__ X, const float* __restrict__ g,
                    const float* __restrict__ bta, bf16* __restrict__ Out) {
    __shared__ float red[9];
    const int D = 1024;
    size_t row = blockIdx.x;
    const float* x = X + row * D;
    float v[4];
#pragma unroll
    for (int i = 0; i < 4; i++) v[i] = x[threadIdx.x + i * 256];
    float s = v[0] + v[1] + v[2] + v[3];
    float mu = blk_sum(s, red) * (1.f / D);
    float q = 0.f;
#pragma unroll
    for (int i = 0; i < 4; i++) { float d = v[i] - mu; q += d * d; }
    float var = blk_sum(q, red) * (1.f / D);
    float rs = rsqrtf(var + 1e-6f);
    bf16* o = Out + row * D;
#pragma unroll
    for (int i = 0; i < 4; i++) {
        int c = threadIdx.x + i * 256;
        o[c] = (bf16)((v[i] - mu) * rs * g[c] + bta[c]);
    }
}

__global__ __launch_bounds__(256)
void ln_res_cast_kernel(const float* __restrict__ A, const float* __restrict__ Bres,
                        const float* __restrict__ g, const float* __restrict__ bta,
                        bf16* __restrict__ Out) {
    __shared__ float red[9];
    const int D = 1024;
    size_t row = blockIdx.x;
    const float* a = A + row * D;
    const float* b = Bres + row * D;
    float v[4];
#pragma unroll
    for (int i = 0; i < 4; i++) { int c = threadIdx.x + i * 256; v[i] = a[c] + b[c]; }
    float s = v[0] + v[1] + v[2] + v[3];
    float mu = blk_sum(s, red) * (1.f / D);
    float q = 0.f;
#pragma unroll
    for (int i = 0; i < 4; i++) { float d = v[i] - mu; q += d * d; }
    float var = blk_sum(q, red) * (1.f / D);
    float rs = rsqrtf(var + 1e-6f);
    bf16* o = Out + row * D;
#pragma unroll
    for (int i = 0; i < 4; i++) {
        int c = threadIdx.x + i * 256;
        o[c] = (bf16)((v[i] - mu) * rs * g[c] + bta[c]);
    }
}

__global__ __launch_bounds__(256)
void ln_final_kernel(float* __restrict__ Y, const float* __restrict__ g,
                     const float* __restrict__ bta) {
    __shared__ float red[9];
    const int D = 1024;
    size_t row = blockIdx.x;
    float* y = Y + row * D;
    float v[4];
#pragma unroll
    for (int i = 0; i < 4; i++) v[i] = y[threadIdx.x + i * 256];
    float s = v[0] + v[1] + v[2] + v[3];
    float mu = blk_sum(s, red) * (1.f / D);
    float q = 0.f;
#pragma unroll
    for (int i = 0; i < 4; i++) { float d = v[i] - mu; q += d * d; }
    float var = blk_sum(q, red) * (1.f / D);
    float rs = rsqrtf(var + 1e-6f);
#pragma unroll
    for (int i = 0; i < 4; i++) {
        int c = threadIdx.x + i * 256;
        y[c] = (v[i] - mu) * rs * g[c] + bta[c];
    }
}

// ---------------------------------------------------------------------------
// bf16 WMMA GEMM: C[M,N] = A[M,K] * B[K,N].
// Block tile 128x64, BK=64, 8 waves in 4x2 layout, each wave 32x32
// (4 accumulators -> 8 WMMAs per barrier).
// Double-buffered LDS tiles filled with global_load_async_to_lds_b128:
//   issue tile k+1 right after the barrier, compute tile k while DMA runs.
// Row strides padded to 72 elements (multiple of 8 -> 16B-aligned chunks).
// Epilogues: 0 = bf16; 1 = f32; 2 = +bias, exact GELU, bf16;
//            3 = +bias +bf16-residual, f32.
// ---------------------------------------------------------------------------
template <int EPI>
__global__ __launch_bounds__(256)
void gemm_bf16(const bf16* __restrict__ A, const bf16* __restrict__ Bw,
               void* __restrict__ Cout, const float* __restrict__ bias,
               const bf16* __restrict__ Res, int M, int N, int K) {
    __shared__ bf16 As[2][128][72];   // [buf][m][k], 2 x 18.0 KB
    __shared__ bf16 Bs[2][64][72];    // [buf][k][n], 2 x  9.0 KB

    const int bm = blockIdx.y * 128, bn = blockIdx.x * 64;
    const int tid = threadIdx.x;
    const int wave = tid >> 5, lane = tid & 31;
    const int wr = wave >> 1, wc = wave & 1;     // wave: 32 rows x 32 cols
    const int l16 = lane & 15, lhi = lane >> 4;

    auto issue_tile = [&](int k0, int buf) {
        // A tile 128x64: 1024 16B chunks, 4 per thread (8 chunks per row)
#pragma unroll
        for (int i = 0; i < 4; i++) {
            int ch = tid * 4 + i;
            int r = ch >> 3, c = ch & 7;
            async_ld_b128((unsigned)(size_t)&As[buf][r][c * 8],
                          (const void*)(A + (size_t)(bm + r) * K + k0 + c * 8));
        }
        // B tile 64x64: 512 16B chunks, 2 per thread (8 chunks per row)
#pragma unroll
        for (int i = 0; i < 2; i++) {
            int ch = tid * 2 + i;
            int r = ch >> 3, c = ch & 7;
            async_ld_b128((unsigned)(size_t)&Bs[buf][r][c * 8],
                          (const void*)(Bw + (size_t)(k0 + r) * N + bn + c * 8));
        }
    };

    v8f acc00, acc01, acc10, acc11;
#pragma unroll
    for (int i = 0; i < 8; i++) { acc00[i] = 0.f; acc01[i] = 0.f; acc10[i] = 0.f; acc11[i] = 0.f; }

    const int nk = K >> 6;
    issue_tile(0, 0);                      // prologue: stage tile 0
    for (int kt = 0; kt < nk; kt++) {
        wait_async0();                     // only tile-kt loads outstanding
        __syncthreads();                   // all waves' fills visible; prev tile consumed
        if (kt + 1 < nk) issue_tile((kt + 1) << 6, (kt + 1) & 1);   // overlap with compute
        const int buf = kt & 1;

#pragma unroll
        for (int ks = 0; ks < 2; ks++) {
            v16bf af0, af1;
#pragma unroll
            for (int e = 0; e < 16; e++) {
                int j = e >> 1;
                int kk = ks * 32 + (lhi << 4) + e;
                af0[e] = As[buf][wr * 32 + l16][kk];
                af1[e] = As[buf][wr * 32 + 16 + l16][kk];
            }
            v16bf fb0, fb1;
#pragma unroll
            for (int e = 0; e < 16; e++) {
                int kk = ks * 32 + (lhi << 4) + e;
                fb0[e] = Bs[buf][kk][wc * 32 + l16];
                fb1[e] = Bs[buf][kk][wc * 32 + 16 + l16];
            }
            acc00 = __builtin_amdgcn_wmma_f32_16x16x32_bf16(false, af0, false, fb0,
                                                            (short)0, acc00, false, false);
            acc01 = __builtin_amdgcn_wmma_f32_16x16x32_bf16(false, af0, false, fb1,
                                                            (short)0, acc01, false, false);
            acc10 = __builtin_amdgcn_wmma_f32_16x16x32_bf16(false, af1, false, fb0,
                                                            (short)0, acc10, false, false);
            acc11 = __builtin_amdgcn_wmma_f32_16x16x32_bf16(false, af1, false, fb1,
                                                            (short)0, acc11, false, false);
        }
    }

    const int mA = bm + wr * 32 + (lhi ? 8 : 0);   // rows for af0-based accums
    const int mB = mA + 16;                        // rows for af1-based accums
    const int n0 = bn + wc * 32 + l16;
    const int n1 = n0 + 16;
#pragma unroll
    for (int r = 0; r < 8; r++) {
        int m0 = mA + r, m1 = mB + r;
        float v00 = acc00[r], v01 = acc01[r], v10 = acc10[r], v11 = acc11[r];
        if (EPI == 0) {
            bf16* C = (bf16*)Cout;
            C[(size_t)m0 * N + n0] = (bf16)v00;  C[(size_t)m0 * N + n1] = (bf16)v01;
            C[(size_t)m1 * N + n0] = (bf16)v10;  C[(size_t)m1 * N + n1] = (bf16)v11;
        } else if (EPI == 1) {
            float* C = (float*)Cout;
            C[(size_t)m0 * N + n0] = v00;  C[(size_t)m0 * N + n1] = v01;
            C[(size_t)m1 * N + n0] = v10;  C[(size_t)m1 * N + n1] = v11;
        } else if (EPI == 2) {
            const float k2 = 0.70710678118654752f;
            v00 += bias[n0]; v01 += bias[n1]; v10 += bias[n0]; v11 += bias[n1];
            v00 = 0.5f * v00 * (1.f + erff(v00 * k2));
            v01 = 0.5f * v01 * (1.f + erff(v01 * k2));
            v10 = 0.5f * v10 * (1.f + erff(v10 * k2));
            v11 = 0.5f * v11 * (1.f + erff(v11 * k2));
            bf16* C = (bf16*)Cout;
            C[(size_t)m0 * N + n0] = (bf16)v00;  C[(size_t)m0 * N + n1] = (bf16)v01;
            C[(size_t)m1 * N + n0] = (bf16)v10;  C[(size_t)m1 * N + n1] = (bf16)v11;
        } else {
            v00 += bias[n0] + (float)Res[(size_t)m0 * N + n0];
            v01 += bias[n1] + (float)Res[(size_t)m0 * N + n1];
            v10 += bias[n0] + (float)Res[(size_t)m1 * N + n0];
            v11 += bias[n1] + (float)Res[(size_t)m1 * N + n1];
            float* C = (float*)Cout;
            C[(size_t)m0 * N + n0] = v00;  C[(size_t)m0 * N + n1] = v01;
            C[(size_t)m1 * N + n0] = v10;  C[(size_t)m1 * N + n1] = v11;
        }
    }
}

// ---------------------------------------------------------------------------
// Per-token head-mixing attention: one wave per token.
// scores(16h x 16g) = Q(16x64) * K^T / 8 ; softmax over g ; O = P * V(16x64).
// Output stored in the reference's scrambled head-major reshape layout.
// ---------------------------------------------------------------------------
__global__ __launch_bounds__(256)
void attn_kernel(const bf16* __restrict__ Q, const bf16* __restrict__ Kb,
                 const bf16* __restrict__ V, bf16* __restrict__ O, int T) {
    __shared__ float Pl[8][16][17];
    const int wave = threadIdx.x >> 5, lane = threadIdx.x & 31;
    const int l16 = lane & 15, lhi = lane >> 4;
    const int token = blockIdx.x * 8 + wave;
    const int b = token / T, t = token % T;
    const bf16* q = Q + (size_t)token * 1024;
    const bf16* k = Kb + (size_t)token * 1024;
    const bf16* v = V + (size_t)token * 1024;

    v8f sc;
#pragma unroll
    for (int i = 0; i < 8; i++) sc[i] = 0.f;

#pragma unroll
    for (int ks = 0; ks < 2; ks++) {
        v16bf aq, bk;
#pragma unroll
        for (int e = 0; e < 16; e++) {
            int j = e >> 1;
            int kk = ks * 32 + (lhi << 4) + e;
            aq[e] = q[l16 * 64 + kk];                 // A: row=h(l16), K=d
            int kd = ks * 32 + (lhi << 4) + e;
            bk[e] = k[l16 * 64 + kd];                 // B: N=g(l16), K=d (K^T)
        }
        sc = __builtin_amdgcn_wmma_f32_16x16x32_bf16(false, aq, false, bk,
                                                     (short)0, sc, false, false);
    }

#pragma unroll
    for (int r = 0; r < 8; r++) {                     // row softmax over g
        float s = sc[r] * 0.125f;
        float mx = s;
#pragma unroll
        for (int o = 1; o < 16; o <<= 1) { float u = __shfl_xor(mx, o, 32); mx = fmaxf(mx, u); }
        float e = __expf(s - mx);
        float sm = e;
#pragma unroll
        for (int o = 1; o < 16; o <<= 1) sm += __shfl_xor(sm, o, 32);
        sc[r] = e / sm;
    }

#pragma unroll
    for (int r = 0; r < 8; r++) Pl[wave][(lhi ? 8 : 0) + r][l16] = sc[r];
    __syncthreads();

    v16bf ap;
#pragma unroll
    for (int e = 0; e < 16; e++) {
        int j = e >> 1;
        int kk = (lhi << 4) + e;
        ap[e] = (kk < 16) ? (bf16)Pl[wave][l16][kk] : (bf16)0.f;   // pad K=16..31
    }

#pragma unroll
    for (int ch = 0; ch < 4; ch++) {
        v16bf bv;
#pragma unroll
        for (int e = 0; e < 16; e++) {
            int kd = (lhi << 4) + e;
            bv[e] = (kd < 16) ? v[kd * 64 + ch * 16 + l16] : (bf16)0.f;
        }
        v8f oc;
#pragma unroll
        for (int i = 0; i < 8; i++) oc[i] = 0.f;
        oc = __builtin_amdgcn_wmma_f32_16x16x32_bf16(false, ap, false, bv,
                                                     (short)0, oc, false, false);
#pragma unroll
        for (int r = 0; r < 8; r++) {
            int h = (lhi ? 8 : 0) + r;
            int d = ch * 16 + l16;
            size_t oi = ((size_t)b * T + h * 256 + (t >> 4)) * 1024 + ((t & 15) << 6) + d;
            O[oi] = (bf16)oc[r];
        }
    }
}

// ---------------------------------------------------------------------------
extern "C" void kernel_launch(void* const* d_in, const int* in_sizes, int n_in,
                              void* d_out, int out_size, void* d_ws, size_t ws_size,
                              hipStream_t stream) {
    (void)in_sizes; (void)n_in; (void)out_size; (void)ws_size;

    const float* x     = (const float*)d_in[0];
    const float* wq    = (const float*)d_in[1];
    const float* wk    = (const float*)d_in[2];
    const float* wv    = (const float*)d_in[3];
    const float* wo    = (const float*)d_in[4];
    const float* ln1_g = (const float*)d_in[5];
    const float* ln1_b = (const float*)d_in[6];
    const float* w1    = (const float*)d_in[7];
    const float* b1    = (const float*)d_in[8];
    const float* w2    = (const float*)d_in[9];
    const float* b2    = (const float*)d_in[10];
    const float* ln2_g = (const float*)d_in[11];
    const float* ln2_b = (const float*)d_in[12];

    const int B = 4, T = 4096, D = 1024, Dff = 4096;
    const int M = B * T;                      // 16384
    char* ws = (char*)d_ws;
    const size_t MB = 1024ull * 1024ull;

    bf16* wq_b = (bf16*)(ws + 0 * MB);
    bf16* wk_b = (bf16*)(ws + 2 * MB);
    bf16* wv_b = (bf16*)(ws + 4 * MB);
    bf16* wo_b = (bf16*)(ws + 6 * MB);
    bf16* w1_b = (bf16*)(ws + 8 * MB);
    bf16* w2_b = (bf16*)(ws + 16 * MB);
    // region A (32 MB): x1 -> o -> q2  (disjoint lifetimes)
    bf16* x1_b = (bf16*)(ws + 24 * MB);
    bf16* o_b  = x1_b;
    bf16* q2_b = x1_b;
    // region B (128 MB): q/k/v -> oWo(f32) -> h  (disjoint lifetimes)
    bf16*  q_b  = (bf16*)(ws + 56 * MB);
    bf16*  k_b  = (bf16*)(ws + 88 * MB);
    bf16*  v_b  = (bf16*)(ws + 120 * MB);
    float* oWo  = (float*)(ws + 56 * MB);
    bf16*  h_b  = (bf16*)(ws + 56 * MB);

    cvt_f32_bf16<<<(D * D + 255) / 256, 256, 0, stream>>>(wq, wq_b, D * D);
    cvt_f32_bf16<<<(D * D + 255) / 256, 256, 0, stream>>>(wk, wk_b, D * D);
    cvt_f32_bf16<<<(D * D + 255) / 256, 256, 0, stream>>>(wv, wv_b, D * D);
    cvt_f32_bf16<<<(D * D + 255) / 256, 256, 0, stream>>>(wo, wo_b, D * D);
    cvt_f32_bf16<<<(D * Dff + 255) / 256, 256, 0, stream>>>(w1, w1_b, D * Dff);
    cvt_f32_bf16<<<(Dff * D + 255) / 256, 256, 0, stream>>>(w2, w2_b, Dff * D);

    ln_cast_kernel<<<M, 256, 0, stream>>>(x, ln1_g, ln1_b, x1_b);

    dim3 gDD(D / 64, M / 128);      // 16 x 128
    gemm_bf16<0><<<gDD, 256, 0, stream>>>(x1_b, wq_b, q_b, nullptr, nullptr, M, D, D);
    gemm_bf16<0><<<gDD, 256, 0, stream>>>(x1_b, wk_b, k_b, nullptr, nullptr, M, D, D);
    gemm_bf16<0><<<gDD, 256, 0, stream>>>(x1_b, wv_b, v_b, nullptr, nullptr, M, D, D);

    attn_kernel<<<M / 8, 256, 0, stream>>>(q_b, k_b, v_b, o_b, T);

    gemm_bf16<1><<<gDD, 256, 0, stream>>>(o_b, wo_b, oWo, nullptr, nullptr, M, D, D);

    ln_res_cast_kernel<<<M, 256, 0, stream>>>(oWo, x, ln1_g, ln1_b, q2_b);

    dim3 gFF(Dff / 64, M / 128);    // 64 x 128
    gemm_bf16<2><<<gFF, 256, 0, stream>>>(q2_b, w1_b, h_b, b1, nullptr, M, Dff, D);

    gemm_bf16<3><<<gDD, 256, 0, stream>>>(h_b, w2_b, d_out, b2, q2_b, M, D, Dff);

    ln_final_kernel<<<M, 256, 0, stream>>>((float*)d_out, ln2_g, ln2_b);
}
